// FingerPrinter_61486751809801
// MI455X (gfx1250) — matrix operand order repaired and checked by reading, not compiled, not run on hardware
//
#include <hip/hip_runtime.h>

typedef __attribute__((ext_vector_type(16))) _Float16 v16h;
typedef __attribute__((ext_vector_type(8)))  _Float16 v8h;
typedef __attribute__((ext_vector_type(8)))  float    v8f;

#define LN_EPS 1e-3f

struct ConvParams {
  const _Float16* in;     // activations f16, NHWC
  const _Float16* wp;     // packed weights f16 [Cout][Kpad]
  const float*    bias;
  const float*    gamma;
  const float*    beta;
  _Float16*       out;    // f16, NHWC
  int Hin, Win, Cin;
  int Hout, Wout, Cout;
  int sh, sw, th, tw, ph, pw;   // strides, tap direction, SAME pad begin
  int Kpad;                     // 3*Cin rounded up to 64
};

__device__ __forceinline__ float elu_f(float v) {
  return v > 0.0f ? v : (__expf(v) - 1.0f);
}

// Single-instruction XOR lane exchange: ds_swizzle_b32, group-of-32 mode
// (offset[4:0]=and=0x1f, offset[9:5]=or=0, offset[14:10]=xor=MASK).
template<int MASK>
__device__ __forceinline__ float ds_swz_xor(float v) {
  constexpr int pat = (MASK << 10) | 0x1f;
  return __int_as_float(__builtin_amdgcn_ds_swizzle(__float_as_int(v), pat));
}

#define WMMA_F16(A, B, C) \
  __builtin_amdgcn_wmma_f32_16x16x32_f16(false, (A), false, (B), (short)0, (C), false, false)

// ---------------------------------------------------------------------------
// Fused conv (GEMM via WMMA) + bias + ELU + LayerNorm(channel).
// Workgroup: 256 threads = 8 waves; M-tile = 16*MT output pixels, all Cout.
// Wave w owns NTILES 16-channel column tiles. K-loop: tap-major (no division),
// 64-wide chunks, double-buffered LDS A panel (one barrier per chunk).
// ---------------------------------------------------------------------------
template<int NTILES, int MT, bool SMALL_CIN>
__global__ __launch_bounds__(256) void conv_elu_ln_kernel(ConvParams p) {
  constexpr int ROWS = 16 * MT;     // pixels per workgroup
  constexpr int TPR  = 256 / ROWS;  // threads cooperating on one A row
  constexpr int ELE  = 64 / TPR;    // f16 elements staged per thread per chunk

  __shared__ __attribute__((aligned(32))) _Float16 As[2][ROWS * 72];
  __shared__ float redS[8][32];
  __shared__ float redQ[8][32];
  __shared__ float meanS[32];
  __shared__ float invS[32];

  const int tid   = threadIdx.x;
  const int wv    = tid >> 5;
  const int lane  = tid & 31;
  const int hlf   = lane >> 4;
  const int l16   = lane & 15;
  const int mtile = blockIdx.x;

  v8f acc[NTILES][MT];
  {
    v8f z = {0.f, 0.f, 0.f, 0.f, 0.f, 0.f, 0.f, 0.f};
#pragma unroll
    for (int nt = 0; nt < NTILES; ++nt)
#pragma unroll
      for (int mt = 0; mt < MT; ++mt) acc[nt][mt] = z;
  }

  // ---- A staging map: this thread's pixel row + element slot (divides once) --
  const int fp_prow = tid / TPR;
  const int fp_j    = (tid % TPR) * ELE;
  int fp_n, fp_ho, fp_wo;
  {
    int pix = mtile * ROWS + fp_prow;
    fp_wo   = pix % p.Wout;
    int t   = pix / p.Wout;
    fp_ho   = t % p.Hout;
    fp_n    = t / p.Hout;
  }
  // Per-tap source pointer + validity, hoisted out of the K loop.
  const _Float16* srcT[3];
  bool valT[3];
#pragma unroll
  for (int tap = 0; tap < 3; ++tap) {
    int hi = fp_ho * p.sh + tap * p.th - p.ph;
    int wi = fp_wo * p.sw + tap * p.tw - p.pw;
    bool v = ((unsigned)hi < (unsigned)p.Hin) && ((unsigned)wi < (unsigned)p.Win);
    valT[tap] = v;
    size_t base = v ? ((size_t)((fp_n * p.Hin + hi) * p.Win + wi) * p.Cin + fp_j) : (size_t)0;
    srcT[tap] = p.in + base;
  }

  // ---- Per-wave weight row base pointers (co*Kpad multiply done once) -------
  const _Float16* wbase[NTILES];
#pragma unroll
  for (int nt = 0; nt < NTILES; ++nt) {
    int co = (wv * NTILES + nt) * 16 + l16;
    wbase[nt] = p.wp + (size_t)co * p.Kpad + hlf * 16;
  }

  int buf = 0;
  if (!SMALL_CIN) {
    int kk = 0;
#pragma unroll
    for (int tap = 0; tap < 3; ++tap) {
      const _Float16* s = srcT[tap];
      const bool      v = valT[tap];
      for (int c0 = 0; c0 < p.Cin; c0 += 64) {
        // stage this thread's slice of the 64-wide A chunk
        _Float16* dst = &As[buf][fp_prow * 72 + fp_j];
        if constexpr (ELE == 8) {
          v8h d = {};
          if (v) d = *(const v8h*)(s + c0);
          *(v8h*)dst = d;
        } else {
          unsigned long long d = 0ull;
          if (v) d = *(const unsigned long long*)(s + c0);
          *(unsigned long long*)dst = d;
        }
        __syncthreads();

        // A fragments (contiguous k = 16h + s, same order as B):
        // lanes 0-15: K {0..15}; lanes 16-31: K {16..31}
        v16h a0[MT], a1[MT];
        const _Float16* ar = &As[buf][l16 * 72 + hlf * 16];
#pragma unroll
        for (int mt = 0; mt < MT; ++mt) {
          const _Float16* r0 = ar + mt * 16 * 72;
          v8h lo0 = *(const v8h*)(r0 + 0);
          v8h hi0 = *(const v8h*)(r0 + 8);
          v8h lo1 = *(const v8h*)(r0 + 32);
          v8h hi1 = *(const v8h*)(r0 + 40);
          a0[mt] = __builtin_shufflevector(lo0, hi0, 0,1,2,3,4,5,6,7,8,9,10,11,12,13,14,15);
          a1[mt] = __builtin_shufflevector(lo1, hi1, 0,1,2,3,4,5,6,7,8,9,10,11,12,13,14,15);
        }
#pragma unroll
        for (int nt = 0; nt < NTILES; ++nt) {
          v16h b0 = *(const v16h*)(wbase[nt] + kk);
          v16h b1 = *(const v16h*)(wbase[nt] + kk + 32);
#pragma unroll
          for (int mt = 0; mt < MT; ++mt) {
            acc[nt][mt] = WMMA_F16(a0[mt], b0, acc[nt][mt]);
            acc[nt][mt] = WMMA_F16(a1[mt], b1, acc[nt][mt]);
          }
        }
        kk += 64;
        buf ^= 1;
      }
    }
  } else {
    // First layer (Cin==1): single zero-padded 64-wide chunk, scalar staging.
#pragma unroll
    for (int e = 0; e < (ROWS * 64) / 256; ++e) {
      int idx  = tid + e * 256;
      int prow = idx >> 6;
      int kkk  = idx & 63;
      _Float16 vv = (_Float16)0.0f;
      int tap = kkk / p.Cin;
      int ci  = kkk - tap * p.Cin;
      if (tap < 3) {
        int pix = mtile * ROWS + prow;
        int wo  = pix % p.Wout;
        int t   = pix / p.Wout;
        int ho  = t % p.Hout;
        int n   = t / p.Hout;
        int hi  = ho * p.sh + tap * p.th - p.ph;
        int wi  = wo * p.sw + tap * p.tw - p.pw;
        if ((unsigned)hi < (unsigned)p.Hin && (unsigned)wi < (unsigned)p.Win)
          vv = p.in[(size_t)((n * p.Hin + hi) * p.Win + wi) * p.Cin + ci];
      }
      As[0][prow * 72 + kkk] = vv;
    }
    __syncthreads();
    const _Float16* ar = &As[0][l16 * 72 + hlf * 16];
#pragma unroll
    for (int mt = 0; mt < MT; ++mt) {
      const _Float16* r0 = ar + mt * 16 * 72;
      v8h lo0 = *(const v8h*)(r0 + 0);
      v8h hi0 = *(const v8h*)(r0 + 8);
      v8h lo1 = *(const v8h*)(r0 + 32);
      v8h hi1 = *(const v8h*)(r0 + 40);
      v16h a0 = __builtin_shufflevector(lo0, hi0, 0,1,2,3,4,5,6,7,8,9,10,11,12,13,14,15);
      v16h a1 = __builtin_shufflevector(lo1, hi1, 0,1,2,3,4,5,6,7,8,9,10,11,12,13,14,15);
#pragma unroll
      for (int nt = 0; nt < NTILES; ++nt) {
        v16h b0 = *(const v16h*)(wbase[nt] + 0);
        v16h b1 = *(const v16h*)(wbase[nt] + 32);
        acc[nt][mt] = WMMA_F16(a0, b0, acc[nt][mt]);
        acc[nt][mt] = WMMA_F16(a1, b1, acc[nt][mt]);
      }
    }
  }

  // ---- Epilogue: bias + ELU (in-register) ----------------------------------
#pragma unroll
  for (int nt = 0; nt < NTILES; ++nt) {
    int co   = (wv * NTILES + nt) * 16 + l16;
    float bi = p.bias[co];
#pragma unroll
    for (int mt = 0; mt < MT; ++mt)
#pragma unroll
      for (int r = 0; r < 8; ++r)
        acc[nt][mt][r] = elu_f(acc[nt][mt][r] + bi);
  }

  // ---- LayerNorm stats: 16-lane XOR tree via ds_swizzle (1 instr/step) -----
#pragma unroll
  for (int mt = 0; mt < MT; ++mt) {
#pragma unroll
    for (int r = 0; r < 8; ++r) {
      float s1 = 0.f, s2 = 0.f;
#pragma unroll
      for (int nt = 0; nt < NTILES; ++nt) {
        float v = acc[nt][mt][r];
        s1 += v; s2 += v * v;
      }
      s1 += ds_swz_xor<1>(s1);  s2 += ds_swz_xor<1>(s2);
      s1 += ds_swz_xor<2>(s1);  s2 += ds_swz_xor<2>(s2);
      s1 += ds_swz_xor<4>(s1);  s2 += ds_swz_xor<4>(s2);
      s1 += ds_swz_xor<8>(s1);  s2 += ds_swz_xor<8>(s2);
      if (l16 == 0) {
        redS[wv][mt * 16 + r + hlf * 8] = s1;
        redQ[wv][mt * 16 + r + hlf * 8] = s2;
      }
    }
  }
  __syncthreads();
  if (tid < ROWS) {
    float s1 = 0.f, s2 = 0.f;
#pragma unroll
    for (int w = 0; w < 8; ++w) { s1 += redS[w][tid]; s2 += redQ[w][tid]; }
    float ic   = 1.0f / (float)p.Cout;
    float mean = s1 * ic;
    float var  = s2 * ic - mean * mean;
    meanS[tid] = mean;
    invS[tid]  = rsqrtf(var + LN_EPS);
  }
  __syncthreads();

  float mn[MT][8], iv[MT][8];
#pragma unroll
  for (int mt = 0; mt < MT; ++mt)
#pragma unroll
    for (int r = 0; r < 8; ++r) {
      int m = mt * 16 + r + hlf * 8;
      mn[mt][r] = meanS[m];
      iv[mt][r] = invS[m];
    }

  // ---- Normalize, affine, store f16 (NHWC: offset = pixel*Cout + co) -------
#pragma unroll
  for (int nt = 0; nt < NTILES; ++nt) {
    int co   = (wv * NTILES + nt) * 16 + l16;
    float g  = p.gamma[co];
    float be = p.beta[co];
#pragma unroll
    for (int mt = 0; mt < MT; ++mt)
#pragma unroll
      for (int r = 0; r < 8; ++r) {
        int m = mt * 16 + r + hlf * 8;
        size_t ob = (size_t)(mtile * ROWS + m) * p.Cout + co;
        p.out[ob] = (_Float16)((acc[nt][mt][r] - mn[mt][r]) * iv[mt][r] * g + be);
      }
  }
}

// Pack HWIO f32 weights -> f16 [Cout][Kpad] (k = tap*Cin + ci), zero-padded K.
__global__ void pack_w_kernel(const float* __restrict__ w, _Float16* __restrict__ wp,
                              int Cout, int K, int Kpad, int total) {
  int i = blockIdx.x * 256 + threadIdx.x;
  if (i >= total) return;
  int co = i / Kpad;
  int k  = i - co * Kpad;
  float v = (k < K) ? w[(size_t)k * Cout + co] : 0.0f;
  wp[i] = (_Float16)v;
}

__global__ void f32_to_f16_kernel(const float* __restrict__ x, _Float16* __restrict__ y, int n) {
  int i = blockIdx.x * 256 + threadIdx.x;
  if (i < n) y[i] = (_Float16)x[i];
}

// 128 per-q MLPs (8->32 ELU ->1) + L2 normalize over q. 64 blocks x 128 threads.
__global__ __launch_bounds__(128) void mlp_l2_kernel(
    const _Float16* __restrict__ h, const float* __restrict__ w1,
    const float* __restrict__ b1, const float* __restrict__ w2,
    const float* __restrict__ b2, float* __restrict__ out) {
  int b = blockIdx.x;
  int q = threadIdx.x;
  float s[8];
#pragma unroll
  for (int i = 0; i < 8; ++i) s[i] = (float)h[b * 1024 + q * 8 + i];
  const float* W1 = w1 + q * 8 * 32;
  const float* B1 = b1 + q * 32;
  const float* W2 = w2 + q * 32;
  float e = b2[q];
#pragma unroll 4
  for (int j = 0; j < 32; ++j) {
    float hj = B1[j];
#pragma unroll
    for (int i = 0; i < 8; ++i) hj += s[i] * W1[i * 32 + j];
    hj = elu_f(hj);
    e += hj * W2[j];
  }
  float sq = e * e;
  sq += ds_swz_xor<1>(sq);
  sq += ds_swz_xor<2>(sq);
  sq += ds_swz_xor<4>(sq);
  sq += ds_swz_xor<8>(sq);
  sq += ds_swz_xor<16>(sq);
  __shared__ float ws4[4];
  int wv = threadIdx.x >> 5;
  if ((threadIdx.x & 31) == 0) ws4[wv] = sq;
  __syncthreads();
  float tot = ws4[0] + ws4[1] + ws4[2] + ws4[3];
  float scale = rsqrtf(fmaxf(tot, 1e-12f));
  out[b * 128 + q] = e * scale;
}

// ---------------------------------------------------------------------------
static constexpr int kCH[8]  = {128, 128, 256, 256, 512, 512, 1024, 1024};
static constexpr int kCIN[8] = {1, 128, 128, 256, 256, 512, 512, 1024};
static constexpr int kS0w[8] = {2, 2, 2, 2, 1, 2, 1, 2};  // S0 = (1, sw)

extern "C" void kernel_launch(void* const* d_in, const int* in_sizes, int n_in,
                              void* d_out, int out_size, void* d_ws, size_t ws_size,
                              hipStream_t stream) {
  (void)in_sizes; (void)out_size; (void)ws_size;

  const float* x = (const float*)d_in[0];
  const float *W1a[8], *B1a[8], *G1a[8], *E1a[8];
  const float *W2a[8], *B2a[8], *G2a[8], *E2a[8];
  const float *fcw1, *fcb1, *fcw2, *fcb2;

  if (n_in >= 69) {
    // Per-leaf flattening: x, w1[0..7], b1[0..7], g1, be1, w2, b2, g2, be2, fc*
    for (int i = 0; i < 8; ++i) {
      W1a[i] = (const float*)d_in[1 + i];
      B1a[i] = (const float*)d_in[9 + i];
      G1a[i] = (const float*)d_in[17 + i];
      E1a[i] = (const float*)d_in[25 + i];
      W2a[i] = (const float*)d_in[33 + i];
      B2a[i] = (const float*)d_in[41 + i];
      G2a[i] = (const float*)d_in[49 + i];
      E2a[i] = (const float*)d_in[57 + i];
    }
    fcw1 = (const float*)d_in[65];
    fcb1 = (const float*)d_in[66];
    fcw2 = (const float*)d_in[67];
    fcb2 = (const float*)d_in[68];
  } else {
    // Lists concatenated per dict key.
    const float* w1c = (const float*)d_in[1];
    const float* b1c = (const float*)d_in[2];
    const float* g1c = (const float*)d_in[3];
    const float* e1c = (const float*)d_in[4];
    const float* w2c = (const float*)d_in[5];
    const float* b2c = (const float*)d_in[6];
    const float* g2c = (const float*)d_in[7];
    const float* e2c = (const float*)d_in[8];
    size_t ow1 = 0, ob = 0, ow2 = 0;
    for (int i = 0; i < 8; ++i) {
      W1a[i] = w1c + ow1; ow1 += (size_t)3 * kCIN[i] * kCH[i];
      B1a[i] = b1c + ob; G1a[i] = g1c + ob; E1a[i] = e1c + ob;
      B2a[i] = b2c + ob; G2a[i] = g2c + ob; E2a[i] = e2c + ob;
      ob += (size_t)kCH[i];
      W2a[i] = w2c + ow2; ow2 += (size_t)3 * kCH[i] * kCH[i];
    }
    fcw1 = (const float*)d_in[9];
    fcb1 = (const float*)d_in[10];
    fcw2 = (const float*)d_in[11];
    fcb2 = (const float*)d_in[12];
  }

  // Workspace carve (ping-pong activations + f16 input + packed weights).
  size_t off = 0;
  char* base = (char*)d_ws;
  auto carve = [&](size_t bytes) -> void* {
    void* ptr = base + off;
    off += (bytes + 255) & ~(size_t)255;
    return ptr;
  };
  _Float16* actA = (_Float16*)carve((size_t)64 * 256 * 16 * 128 * 2);  // even layers
  _Float16* actB = (_Float16*)carve((size_t)64 * 128 * 16 * 128 * 2);  // odd layers
  _Float16* x16  = (_Float16*)carve((size_t)64 * 256 * 32 * 2);

  int nx = 64 * 256 * 32;
  hipLaunchKernelGGL(f32_to_f16_kernel, dim3((nx + 255) / 256), dim3(256), 0, stream,
                     x, x16, nx);

  const _Float16* cur = x16;
  int Hc = 256, Wc = 32;
  for (int i = 0; i < 8; ++i) {
    for (int j = 0; j < 2; ++j) {
      ConvParams p;
      p.Hin = Hc; p.Win = Wc;
      const float* wsrc;
      if (j == 0) {  // conv1x3, stride (1, sw), taps along W
        p.Cin = kCIN[i]; p.Cout = kCH[i];
        p.sh = 1; p.sw = kS0w[i]; p.th = 0; p.tw = 1;
        p.bias = B1a[i]; p.gamma = G1a[i]; p.beta = E1a[i];
        wsrc = W1a[i];
      } else {       // conv3x1, stride (2, 1), taps along H
        p.Cin = kCH[i]; p.Cout = kCH[i];
        p.sh = 2; p.sw = 1; p.th = 1; p.tw = 0;
        p.bias = B2a[i]; p.gamma = G2a[i]; p.beta = E2a[i];
        wsrc = W2a[i];
      }
      p.Hout = (Hc + p.sh - 1) / p.sh;
      p.Wout = (Wc + p.sw - 1) / p.sw;
      int kh = p.th ? 3 : 1, kw = p.tw ? 3 : 1;
      int pth = (p.Hout - 1) * p.sh + kh - Hc; if (pth < 0) pth = 0;
      int ptw = (p.Wout - 1) * p.sw + kw - Wc; if (ptw < 0) ptw = 0;
      p.ph = pth / 2; p.pw = ptw / 2;          // TF SAME: pad_beg = total/2
      int Ktrue = 3 * p.Cin;
      p.Kpad = (Ktrue + 63) & ~63;

      _Float16* wp = (_Float16*)carve((size_t)p.Cout * p.Kpad * 2);
      int tot = p.Cout * p.Kpad;
      hipLaunchKernelGGL(pack_w_kernel, dim3((tot + 255) / 256), dim3(256), 0, stream,
                         wsrc, wp, p.Cout, Ktrue, p.Kpad, tot);
      p.wp = wp;
      p.in = cur;
      _Float16* outbuf = (((2 * i + j) & 1) == 0) ? actA : actB;
      p.out = outbuf;

      int M = 64 * p.Hout * p.Wout;
      int nt = p.Cout / 128;
      bool small = (p.Cin % 64) != 0;
      if (small)
        hipLaunchKernelGGL((conv_elu_ln_kernel<1, 2, true>),  dim3(M / 32), dim3(256), 0, stream, p);
      else if (nt == 1)
        hipLaunchKernelGGL((conv_elu_ln_kernel<1, 2, false>), dim3(M / 32), dim3(256), 0, stream, p);
      else if (nt == 2)
        hipLaunchKernelGGL((conv_elu_ln_kernel<2, 2, false>), dim3(M / 32), dim3(256), 0, stream, p);
      else if (nt == 4)
        hipLaunchKernelGGL((conv_elu_ln_kernel<4, 2, false>), dim3(M / 32), dim3(256), 0, stream, p);
      else  // Cout == 1024: keep MT=1 to bound accumulator VGPRs (64 + frags)
        hipLaunchKernelGGL((conv_elu_ln_kernel<8, 1, false>), dim3(M / 16), dim3(256), 0, stream, p);

      cur = outbuf; Hc = p.Hout; Wc = p.Wout;
    }
  }

  hipLaunchKernelGGL(mlp_l2_kernel, dim3(64), dim3(128), 0, stream,
                     cur, fcw1, fcb1, fcw2, fcb2, (float*)d_out);
}
